// PhasorBlock_71511205478672
// MI455X (gfx1250) — matrix operand order repaired and checked by reading, not compiled, not run on hardware
//
#include <hip/hip_runtime.h>
#include <math.h>
#include <stdint.h>

typedef __attribute__((ext_vector_type(16))) _Float16 v16h;
typedef __attribute__((ext_vector_type(8)))  float    v8f;
typedef __attribute__((ext_vector_type(4)))  int      v4i;

#define WMMA_F16(a,b,c) __builtin_amdgcn_wmma_f32_16x16x32_f16(false,(a),false,(b),(short)0,(c),false,false)

#if __has_builtin(__builtin_amdgcn_global_load_async_to_lds_b128)
#define HAS_ASYNC_LDS 1
#else
#define HAS_ASYNC_LDS 0
#endif

__device__ __forceinline__ void wait_async_zero()
{
#if __has_builtin(__builtin_amdgcn_s_wait_asynccnt)
  __builtin_amdgcn_s_wait_asynccnt(0);
#elif HAS_ASYNC_LDS
  asm volatile("s_wait_asynccnt 0x0" ::: "memory");
#endif
}

constexpr int L = 2048, D = 256, P = 32, NH = 4, WN = 5;
constexpr int T = 64, CCH = L / T;               // chunk size / #chunks
constexpr float PI_F  = 3.14159265358979323846f;
constexpr float SQRTP = 5.65685424949238f;       // sqrt(P)

__device__ __forceinline__ float gelu_f(float v){ return 0.5f*v*(1.0f+erff(v*0.70710678118654752f)); }
__device__ __forceinline__ float sigm_f(float v){ return 1.0f/(1.0f+expf(-v)); }

// Load one ISA-layout A fragment (16x32 f16) for this lane from a row pointer
// already offset to k0. Lane layout: m = lane&15, k = (i<8?i:i+8)+8*hi.
// Each half-fragment is 8 contiguous floats -> 2x float4 vector loads.
__device__ __forceinline__ v16h load_a_frag(const float* __restrict__ rowk0, int hi)
{
  const float4* q = (const float4*)(rowk0 + 8*hi);
  const float4 x0 = q[0], x1 = q[1];   // k = k0+8hi .. +7
  const float4 y0 = q[4], y1 = q[5];   // k = k0+16+8hi .. +7
  v16h a;
  a[0]=(_Float16)x0.x;  a[1]=(_Float16)x0.y;  a[2]=(_Float16)x0.z;  a[3]=(_Float16)x0.w;
  a[4]=(_Float16)x1.x;  a[5]=(_Float16)x1.y;  a[6]=(_Float16)x1.z;  a[7]=(_Float16)x1.w;
  a[8]=(_Float16)y0.x;  a[9]=(_Float16)y0.y;  a[10]=(_Float16)y0.z; a[11]=(_Float16)y0.w;
  a[12]=(_Float16)y1.x; a[13]=(_Float16)y1.y; a[14]=(_Float16)y1.z; a[15]=(_Float16)y1.w;
  return a;
}

// ---------------------------------------------------------------------------
// Generic WMMA GEMM: C[M,N] = act(A[M,K] @ B[K,N] + bias). One wave per 16x16
// tile. K, activation, N-bounds-checking are compile-time. M is grid-exact.
// ---------------------------------------------------------------------------
template<int K, int ACT, bool NB>
__global__ void gemm_wmma(const float* __restrict__ A, const float* __restrict__ B,
                          const float* __restrict__ bias, float* __restrict__ Cout, int N)
{
  const int lane = threadIdx.x & 31;
  const int m16  = lane & 15;
  const int hi   = lane >> 4;
  const int tn = blockIdx.x, tm = blockIdx.y;
  const int arow = tm*16 + m16;
  const int bcol = tn*16 + m16;
  const float* __restrict__ Arow = A + (size_t)arow*K;
  const bool colok = !NB || (bcol < N);
  __builtin_prefetch(Arow, 0, 1);
  v8f c = {};
#pragma unroll
  for (int k0 = 0; k0 < K; k0 += 32) {
    const v16h a = load_a_frag(Arow + k0, hi);
    v16h b;
#pragma unroll
    for (int i = 0; i < 16; ++i) {
      const int kb = k0 + (i < 8 ? i : i + 8) + 8*hi;
      b[i] = colok ? (_Float16)B[(size_t)kb*N + bcol] : (_Float16)0.f;
    }
    c = WMMA_F16(a, b, c);
  }
#pragma unroll
  for (int j = 0; j < 8; ++j) {
    const int r = tm*16 + j + 8*hi;
    if (colok) {
      float v = c[j] + bias[bcol];
      if (ACT == 1) v = gelu_f(v);
      else if (ACT == 2) v = sigm_f(v);
      Cout[(size_t)r*N + bcol] = v;
    }
  }
}

// ---------------------------------------------------------------------------
// Chunked causal linear attention, phase 1: per-chunk state sums
// Ssum[ch][j][d] = sum_t Kcat[l0+t][j] * V[l0+t][d]   (j:64, d:256, t:T)
// ---------------------------------------------------------------------------
__global__ void chunk_outer(const float* __restrict__ Kc, const float* __restrict__ V,
                            float* __restrict__ Ssum)
{
  const int lane = threadIdx.x & 31, m16 = lane & 15, hi = lane >> 4;
  const int tn = blockIdx.x;            // d-tile 0..15
  const int tm = blockIdx.y;            // j-tile 0..3
  const int ch = blockIdx.z;
  const int l0 = ch * T;
  v8f c = {};
#pragma unroll
  for (int k0 = 0; k0 < T; k0 += 32) {
    v16h a, b;
#pragma unroll
    for (int i = 0; i < 16; ++i) {
      const int ta = k0 + (i < 8 ? i : i + 8) + 8*hi;
      a[i] = (_Float16)Kc[(size_t)(l0 + ta)*64 + tm*16 + m16];
      const int tb = k0 + (i < 8 ? i : i + 8) + 8*hi;
      b[i] = (_Float16)V[(size_t)(l0 + tb)*D + tn*16 + m16];
    }
    c = WMMA_F16(a, b, c);
  }
#pragma unroll
  for (int j = 0; j < 8; ++j)
    Ssum[((size_t)ch*64 + tm*16 + j + 8*hi)*D + tn*16 + m16] = c[j];
}

// phase 2: exclusive prefix over chunks, 16K independent lanes, 32 serial steps
__global__ void prefix_chunks(float* __restrict__ S, int nch, int stride)
{
  const int i = blockIdx.x*blockDim.x + threadIdx.x;
  if (i >= stride) return;
  float run = 0.f;
  for (int ch = 0; ch < nch; ++ch) {
    const float t = S[(size_t)ch*stride + i];
    S[(size_t)ch*stride + i] = run;
    run += t;
  }
}

// ---------------------------------------------------------------------------
// phase 3: per-chunk  Out[l] = Q[l] @ Spref[ch]  +  (masked QK^T)[l,:] @ V
// 128 threads = 4 waves; wave w owns rows l_local = w*16 .. w*16+15.
// Key tile is staged into LDS with GLOBAL_LOAD_ASYNC_TO_LDS (ASYNCcnt) when
// the toolchain exposes the builtin; values tile is converted to f16 in LDS.
// ---------------------------------------------------------------------------
__global__ void causal_chunk(const float* __restrict__ Q, const float* __restrict__ Kc,
                             const float* __restrict__ V, const float* __restrict__ Spref,
                             float* __restrict__ Out)
{
  __shared__ float    sK32[T][64];      // keys chunk [t][j] (f32, async-DMA'd)
  __shared__ _Float16 sV[T][D];         // values chunk [t][d] (f16)
  __shared__ _Float16 sS[4][16][T];     // masked scores per wave [r][t]
  const int tid  = threadIdx.x;
  const int wave = tid >> 5, lane = tid & 31, m16 = lane & 15, hi = lane >> 4;
  const int ch = blockIdx.x;
  const int l0 = ch * T;

#if HAS_ASYNC_LDS
  for (int idx = tid; idx < T*64/4; idx += 128) {      // 16B per async op
    const int t = idx >> 4, j = (idx & 15) * 4;
    const float* src = &Kc[(size_t)(l0 + t)*64 + j];
    __builtin_amdgcn_global_load_async_to_lds_b128(
        (v4i*)(uintptr_t)src,
        (__attribute__((address_space(3))) v4i*)(uintptr_t)&sK32[t][j],
        0, 0);
  }
#else
  for (int idx = tid; idx < T*64/4; idx += 128) {
    const int t = idx >> 4, j = (idx & 15) * 4;
    *(float4*)&sK32[t][j] = *(const float4*)&Kc[(size_t)(l0 + t)*64 + j];
  }
#endif
  for (int idx = tid; idx < T*D/4; idx += 128) {       // values: float4 -> 4x f16
    const int t = idx >> 6, d = (idx & 63) * 4;
    const float4 v = *(const float4*)&V[(size_t)(l0 + t)*D + d];
    sV[t][d+0]=(_Float16)v.x; sV[t][d+1]=(_Float16)v.y;
    sV[t][d+2]=(_Float16)v.z; sV[t][d+3]=(_Float16)v.w;
  }
  wait_async_zero();
  __syncthreads();

  // Q A-fragments (16x64, two k-steps) for this wave's rows — vector loads
  const float* __restrict__ Qrow = Q + (size_t)(l0 + wave*16 + m16)*64;
  v16h aq[2];
#pragma unroll
  for (int ks = 0; ks < 2; ++ks) aq[ks] = load_a_frag(Qrow + ks*32, hi);

  // scores S[r][t] = Q[r]·K[t], causal mask t <= l_local, store f16
#pragma unroll
  for (int st = 0; st < 4; ++st) {
    v8f s = {};
#pragma unroll
    for (int ks = 0; ks < 2; ++ks) {
      v16h b;
#pragma unroll
      for (int i = 0; i < 16; ++i) {
        const int k = ks*32 + (i < 8 ? i : i + 8) + 8*hi;
        b[i] = (_Float16)sK32[st*16 + m16][k];
      }
      s = WMMA_F16(aq[ks], b, s);
    }
    const int tcol = st*16 + m16;
#pragma unroll
    for (int j = 0; j < 8; ++j) {
      const int r = j + 8*hi;
      sS[wave][r][tcol] = (tcol <= wave*16 + r) ? (_Float16)s[j] : (_Float16)0.f;
    }
  }
  __syncthreads();

  // intra A-fragments from masked scores (16x64)
  v16h as[2];
#pragma unroll
  for (int ks = 0; ks < 2; ++ks)
#pragma unroll
    for (int i = 0; i < 16; ++i) {
      const int t = ks*32 + (i < 8 ? i : i + 8) + 8*hi;
      as[ks][i] = sS[wave][m16][t];
    }

  for (int dt = 0; dt < 16; ++dt) {
    v8f acc = {};
#pragma unroll
    for (int ks = 0; ks < 2; ++ks) {                 // inter: Q @ Spref
      v16h b;
#pragma unroll
      for (int i = 0; i < 16; ++i) {
        const int k = ks*32 + (i < 8 ? i : i + 8) + 8*hi;
        b[i] = (_Float16)Spref[((size_t)ch*64 + k)*D + dt*16 + m16];
      }
      acc = WMMA_F16(aq[ks], b, acc);
    }
#pragma unroll
    for (int ks = 0; ks < 2; ++ks) {                 // intra: S @ V
      v16h b;
#pragma unroll
      for (int i = 0; i < 16; ++i) {
        const int t = ks*32 + (i < 8 ? i : i + 8) + 8*hi;
        b[i] = sV[t][dt*16 + m16];
      }
      acc = WMMA_F16(as[ks], b, acc);
    }
#pragma unroll
    for (int j = 0; j < 8; ++j) {
      const int l = l0 + wave*16 + j + 8*hi;
      Out[(size_t)l*D + dt*16 + m16] = acc[j];
    }
  }
}

// ------------------------------ elementwise --------------------------------
__global__ void k_poskcat(const float* __restrict__ posph, float* __restrict__ katA)
{
  const int idx = blockIdx.x*blockDim.x + threadIdx.x;
  if (idx >= L*P) return;
  const int l = idx / P, p = idx % P;
  const float ph = posph[(size_t)l*P + p];
  katA[(size_t)l*64 + p]      = cosf(ph);
  katA[(size_t)l*64 + 32 + p] = sinf(ph);
}

__global__ void k_offin(const float* __restrict__ x, const float* __restrict__ posph,
                        float* __restrict__ off_in)
{
  const int idx = blockIdx.x*blockDim.x + threadIdx.x;
  if (idx >= L*(D+P)) return;
  const int l = idx / (D+P), d = idx % (D+P);
  off_in[idx] = (d < D) ? x[(size_t)l*D + d] : posph[(size_t)l*P + (d - D)];
}

// collapse H heads x WN windows into one effective 64-dim query (linear in q)
__global__ void k_qeff(const float* __restrict__ ao, const float* __restrict__ katA,
                       const float* __restrict__ deltas, float* __restrict__ qatA)
{
  const int l = blockIdx.x*blockDim.x + threadIdx.x;
  if (l >= L) return;
  float qe[64];
#pragma unroll
  for (int i = 0; i < 64; ++i) qe[i] = 0.f;
  float pc[P], ps[P];
#pragma unroll
  for (int p = 0; p < P; ++p) { pc[p] = katA[(size_t)l*64 + p]; ps[p] = katA[(size_t)l*64 + 32 + p]; }
  for (int h = 0; h < NH; ++h) {
    const float* a = &ao[(size_t)l*(NH*(P+1)) + h*(P+1)];
    float ctr[P];
    for (int p = 0; p < P; ++p) ctr[p] = tanhf(a[p]) * PI_F;
    const float wd = sigm_f(a[P]) * 2.f + 0.1f;
    float wts[WN]; float den = 1e-6f;
    for (int w = 0; w < WN; ++w) {
      const float dl = deltas[w];
      wts[w] = expf(-0.5f * dl*dl / (wd*wd + 1e-6f));
      den += wts[w];
    }
    for (int w = 0; w < WN; ++w) {
      const float coef = wts[w] / (den * (float)NH * SQRTP);
      const float dph  = deltas[w] * 0.1f;
      for (int p = 0; p < P; ++p) {
        const float op = ctr[p] + dph;
        const float oc = cosf(op), os = sinf(op);
        qe[p]      += coef * (pc[p]*oc - ps[p]*os);
        qe[32 + p] += coef * (ps[p]*oc + pc[p]*os);
      }
    }
  }
#pragma unroll
  for (int i = 0; i < 64; ++i) qatA[(size_t)l*64 + i] = qe[i];
}

__global__ void k_phasecat(const float* __restrict__ pre, float* __restrict__ cat)
{
  const int idx = blockIdx.x*blockDim.x + threadIdx.x;
  if (idx >= L*P) return;
  const int l = idx / P, p = idx % P;
  const float ph = tanhf(pre[(size_t)l*P + p]) * PI_F;
  cat[(size_t)l*64 + p]      = cosf(ph);
  cat[(size_t)l*64 + 32 + p] = sinf(ph);
}

__global__ void k_gv(const float* __restrict__ values, const float* __restrict__ gate,
                     float* __restrict__ gv)
{
  const int idx = blockIdx.x*blockDim.x + threadIdx.x;
  if (idx >= L*D) return;
  gv[idx] = values[idx] * gate[idx >> 8];
}

// ctx cumsum per column + build storage_in = [x | ctx/(l+1)]
__global__ void k_ctx_storage(const float* __restrict__ kce, const float* __restrict__ x,
                              float* __restrict__ stor)
{
  const int d = threadIdx.x;
  float run = 0.f;
  for (int l = 0; l < L; ++l) {
    run += kce[(size_t)l*D + d];
    stor[(size_t)l*2*D + d]     = x[(size_t)l*D + d];
    stor[(size_t)l*2*D + D + d] = run / (float)(l + 1);
  }
}

__global__ void k_gatecum(const float* __restrict__ gate, float* __restrict__ gcum)
{
  if (threadIdx.x != 0 || blockIdx.x != 0) return;
  float run = 0.f;
  for (int l = 0; l < L; ++l) { run += gate[l]; gcum[l] = fmaxf(run, 1.0f); }
}

__global__ void k_combine_ln(const float* __restrict__ mh, const float* __restrict__ kv,
                             const float* __restrict__ gcum, const float* __restrict__ ln_g,
                             const float* __restrict__ ln_b, float* __restrict__ lnx)
{
  __shared__ float red[D];
  const int l = blockIdx.x, d = threadIdx.x;
  const float scale = 1.0f / (sqrtf(gcum[l]) * SQRTP);
  const float c = mh[(size_t)l*D + d] + kv[(size_t)l*D + d] * scale;
  red[d] = c; __syncthreads();
  for (int s = D/2; s > 0; s >>= 1) { if (d < s) red[d] += red[d+s]; __syncthreads(); }
  const float mu = red[0] / (float)D; __syncthreads();
  red[d] = (c - mu) * (c - mu); __syncthreads();
  for (int s = D/2; s > 0; s >>= 1) { if (d < s) red[d] += red[d+s]; __syncthreads(); }
  const float var = red[0] / (float)D;
  lnx[(size_t)l*D + d] = (c - mu) * rsqrtf(var + 1e-5f) * ln_g[d] + ln_b[d];
}

__global__ void k_final(const float* __restrict__ x, const float* __restrict__ tmp,
                        float* __restrict__ out)
{
  const int idx = blockIdx.x*blockDim.x + threadIdx.x;
  if (idx >= L*D) return;
  out[idx] = x[idx] + tmp[idx];
}

// ---------------------------------------------------------------------------
extern "C" void kernel_launch(void* const* d_in, const int* in_sizes, int n_in,
                              void* d_out, int out_size, void* d_ws, size_t ws_size,
                              hipStream_t stream)
{
  (void)in_sizes; (void)n_in; (void)out_size; (void)ws_size;
  const float* x      = (const float*)d_in[0];
  const float* posph  = (const float*)d_in[1];
  const float* deltas = (const float*)d_in[2];
  const float* mc_w   = (const float*)d_in[3];  const float* mc_b   = (const float*)d_in[4];
  const float* hop_w1 = (const float*)d_in[5];  const float* hop_b1 = (const float*)d_in[6];
  const float* hop_w2 = (const float*)d_in[7];  const float* hop_b2 = (const float*)d_in[8];
  const float* cpe_w  = (const float*)d_in[9];  const float* cpe_b  = (const float*)d_in[10];
  const float* ve_w   = (const float*)d_in[11]; const float* ve_b   = (const float*)d_in[12];
  const float* kce_w  = (const float*)d_in[13]; const float* kce_b  = (const float*)d_in[14];
  const float* ksk_w1 = (const float*)d_in[15]; const float* ksk_b1 = (const float*)d_in[16];
  const float* ksk_w2 = (const float*)d_in[17]; const float* ksk_b2 = (const float*)d_in[18];
  const float* sg_w1  = (const float*)d_in[19]; const float* sg_b1  = (const float*)d_in[20];
  const float* sg_w2  = (const float*)d_in[21]; const float* sg_b2  = (const float*)d_in[22];
  const float* ln_g   = (const float*)d_in[23]; const float* ln_b   = (const float*)d_in[24];
  const float* out_w  = (const float*)d_in[25]; const float* out_b  = (const float*)d_in[26];
  float* out = (float*)d_out;

  float* w = (float*)d_ws;
  auto alloc = [&](size_t n) { float* p = w; w += n; return p; };
  float* katA   = alloc((size_t)L*64);
  float* qatA   = alloc((size_t)L*64);
  float* content= alloc((size_t)L*D);
  float* off_in = alloc((size_t)L*(D+P));
  float* h1     = alloc((size_t)L*D);
  float* ao     = alloc((size_t)L*NH*(P+1));
  float* cpepre = alloc((size_t)L*P);
  float* qatB   = alloc((size_t)L*64);
  float* values = alloc((size_t)L*D);
  float* kce    = alloc((size_t)L*D);
  float* sg1    = alloc((size_t)L*(D/2));
  float* gate   = alloc((size_t)L);
  float* gcum   = alloc((size_t)L);
  float* stor   = alloc((size_t)L*2*D);
  float* s1     = alloc((size_t)L*D);
  float* stpre  = alloc((size_t)L*P);
  float* katB   = alloc((size_t)L*64);
  float* gv     = alloc((size_t)L*D);
  float* ssumA  = alloc((size_t)CCH*64*D);
  float* ssumB  = alloc((size_t)CCH*64*D);
  float* mh     = alloc((size_t)L*D);
  float* kv     = alloc((size_t)L*D);
  float* lnx    = alloc((size_t)L*D);
  float* tmp    = alloc((size_t)L*D);

  const dim3 blk32(32);
  // positional phasor bank
  k_poskcat<<<dim3((L*P + 255)/256), 256, 0, stream>>>(posph, katA);
  // content = x @ mc_w + mc_b
  gemm_wmma<256,0,false><<<dim3(D/16, L/16), blk32, 0, stream>>>(x, mc_w, mc_b, content, D);
  // hop MLP -> ao -> effective query
  k_offin<<<dim3((L*(D+P) + 255)/256), 256, 0, stream>>>(x, posph, off_in);
  gemm_wmma<288,1,false><<<dim3(D/16, L/16), blk32, 0, stream>>>(off_in, hop_w1, hop_b1, h1, D);
  gemm_wmma<256,0,true><<<dim3((NH*(P+1) + 15)/16, L/16), blk32, 0, stream>>>(h1, hop_w2, hop_b2, ao, NH*(P+1));
  k_qeff<<<dim3(L/64), 64, 0, stream>>>(ao, katA, deltas, qatA);
  // scan A (phasor memory) as chunked causal linear attention
  chunk_outer<<<dim3(D/16, 4, CCH), blk32, 0, stream>>>(katA, content, ssumA);
  prefix_chunks<<<dim3((64*D + 255)/256), 256, 0, stream>>>(ssumA, CCH, 64*D);
  causal_chunk<<<dim3(CCH), 128, 0, stream>>>(qatA, katA, content, ssumA, mh);
  // key/value branch
  gemm_wmma<256,0,false><<<dim3(P/16, L/16), blk32, 0, stream>>>(x, cpe_w, cpe_b, cpepre, P);
  k_phasecat<<<dim3((L*P + 255)/256), 256, 0, stream>>>(cpepre, qatB);
  gemm_wmma<256,0,false><<<dim3(D/16, L/16), blk32, 0, stream>>>(x, ve_w, ve_b, values, D);
  gemm_wmma<256,0,false><<<dim3(D/16, L/16), blk32, 0, stream>>>(x, kce_w, kce_b, kce, D);
  gemm_wmma<256,1,false><<<dim3((D/2)/16, L/16), blk32, 0, stream>>>(x, sg_w1, sg_b1, sg1, D/2);
  gemm_wmma<128,2,true><<<dim3(1, L/16), blk32, 0, stream>>>(sg1, sg_w2, sg_b2, gate, 1);
  k_ctx_storage<<<dim3(1), D, 0, stream>>>(kce, x, stor);
  gemm_wmma<512,1,false><<<dim3(D/16, L/16), blk32, 0, stream>>>(stor, ksk_w1, ksk_b1, s1, D);
  gemm_wmma<256,0,false><<<dim3(P/16, L/16), blk32, 0, stream>>>(s1, ksk_w2, ksk_b2, stpre, P);
  k_phasecat<<<dim3((L*P + 255)/256), 256, 0, stream>>>(stpre, katB);
  k_gv<<<dim3((L*D + 255)/256), 256, 0, stream>>>(values, gate, gv);
  // scan B (kv memory)
  chunk_outer<<<dim3(D/16, 4, CCH), blk32, 0, stream>>>(katB, gv, ssumB);
  prefix_chunks<<<dim3((64*D + 255)/256), 256, 0, stream>>>(ssumB, CCH, 64*D);
  causal_chunk<<<dim3(CCH), 128, 0, stream>>>(qatB, katB, gv, ssumB, kv);
  k_gatecum<<<dim3(1), 1, 0, stream>>>(gate, gcum);
  // combine + layernorm + output projection + residual
  k_combine_ln<<<dim3(L), D, 0, stream>>>(mh, kv, gcum, ln_g, ln_b, lnx);
  gemm_wmma<256,0,false><<<dim3(D/16, L/16), blk32, 0, stream>>>(lnx, out_w, out_b, tmp, D);
  k_final<<<dim3((L*D + 255)/256), 256, 0, stream>>>(x, tmp, out);
}
